// GraphConv_KGAT_58067957842411
// MI455X (gfx1250) — matrix operand order repaired and checked by reading, not compiled, not run on hardware
//
#include <hip/hip_runtime.h>
#include <hip/hip_bf16.h>

#define N_NODES  50000
#define N_EDGES  800000
#define IN_CH    128
#define OUT_CH   64
#define NEG_SLOPE 0.2f
#define E_TOT    (N_EDGES + N_NODES)
#define ROW_TILES ((N_NODES + 15) / 16)   /* 3125, exact: 50000 = 3125*16 */
#define KSTEPS   (IN_CH / 32)             /* 4 */

typedef __attribute__((ext_vector_type(16))) __bf16 v16bf;
typedef __attribute__((ext_vector_type(8)))  float  v8f;

// fp32 -> bf16 bits, round-to-nearest-even
__device__ __forceinline__ unsigned bfbits(float f) {
  union { float f; unsigned u; } v; v.f = f;
  return (v.u + 0x7FFFu + ((v.u >> 16) & 1u)) >> 16;
}
__device__ __forceinline__ unsigned pack2bf(float lo, float hi) {
  return bfbits(lo) | (bfbits(hi) << 16);
}

// order-preserving float<->uint key for atomicMax-based segment max
__device__ __forceinline__ unsigned fkey(float f) {
  unsigned u = __float_as_uint(f);
  return (u & 0x80000000u) ? ~u : (u | 0x80000000u);
}
__device__ __forceinline__ float funkey(unsigned k) {
  unsigned u = (k & 0x80000000u) ? (k ^ 0x80000000u) : ~k;
  return __uint_as_float(u);
}

// --------------------------------------------------------------------------
// Pre-pass A: pack x (fp32 row-major) into WMMA A-fragment layout (bf16).
// Dword index: (((rowTile*KSTEPS) + kk4)*32 + lane)*8 + v
// lane=(g,m): v<4 -> K = 32*kk4 + 8g + 2v(+1); v>=4 -> K = 32*kk4 + 16 + 8g + 2(v-4)(+1)
// --------------------------------------------------------------------------
__global__ void gat_pack_a(const float* __restrict__ x, unsigned* __restrict__ apk) {
  const long t = (long)blockIdx.x * blockDim.x + threadIdx.x;
  if (t >= (long)ROW_TILES * KSTEPS * 32 * 8) return;
  const int v    = (int)(t & 7);
  const int lane = (int)((t >> 3) & 31);
  const int kk4  = (int)((t >> 8) & (KSTEPS - 1));
  const int rowTile = (int)(t >> 10);
  const int g = lane >> 4, m = lane & 15;
  const int row = rowTile * 16 + m;
  const int k0 = kk4 * 32 + ((v < 4) ? (8 * g + 2 * v) : (16 + 8 * g + 2 * (v - 4)));
  const float* xr = x + (long)row * IN_CH + k0;
  apk[t] = pack2bf(xr[0], xr[1]);
}

// --------------------------------------------------------------------------
// Pre-pass B: pack W (K=128 x N=64 row-major fp32) into B-fragment layout.
// Dword index: ((kk4*4 + ct)*32 + lane)*8 + v
// lane=(g,n): v<4 -> K = 32*kk4 + 8g + 2v(+1); v>=4 -> K = 32*kk4 + 16 + 8g + 2(v-4)(+1), col = ct*16 + n
// --------------------------------------------------------------------------
__global__ void gat_pack_b(const float* __restrict__ w, unsigned* __restrict__ bpk) {
  const int t = blockIdx.x * blockDim.x + threadIdx.x;
  if (t >= KSTEPS * 4 * 32 * 8) return;
  const int v    = t & 7;
  const int lane = (t >> 3) & 31;
  const int ct   = (t >> 8) & 3;
  const int kk4  = t >> 10;
  const int g = lane >> 4, n = lane & 15;
  const int col = ct * 16 + n;
  const int k0  = kk4 * 32 + ((v < 4) ? (8 * g + 2 * v) : (16 + 8 * g + 2 * (v - 4)));
  bpk[t] = pack2bf(w[k0 * OUT_CH + col], w[(k0 + 1) * OUT_CH + col]);
}

// --------------------------------------------------------------------------
// Kernel 1: h = x @ W. Fragments pre-packed -> pure b128 loads + v_wmma.
// One wave per 16-row tile; 4 col-tiles of 16; K-loop 4 x 32.
// --------------------------------------------------------------------------
__global__ void gat_gemm_bf16(const v16bf* __restrict__ apk,
                              const v16bf* __restrict__ bpk,
                              float* __restrict__ h) {
  const int lane = threadIdx.x & 31;
  const int wave = threadIdx.x >> 5;
  const int rowTile = blockIdx.x * (blockDim.x >> 5) + wave;
  if (rowTile >= ROW_TILES) return;          // uniform per wave -> EXEC stays all-1
  const int g = lane >> 4;
  const int m = lane & 15;

  v8f acc[4] = {};

  #pragma unroll
  for (int kk4 = 0; kk4 < KSTEPS; ++kk4) {
    const v16bf a = apk[((long)rowTile * KSTEPS + kk4) * 32 + lane];
    #pragma unroll
    for (int ct = 0; ct < 4; ++ct) {
      const v16bf b = bpk[(kk4 * 4 + ct) * 32 + lane];
      acc[ct] = __builtin_amdgcn_wmma_f32_16x16x32_bf16(
          false, a, false, b, (short)0, acc[ct], false, false);
    }
  }

  // D layout: VGPR r, lane (g,m) = D[r + 8g][m]
  #pragma unroll
  for (int ct = 0; ct < 4; ++ct) {
    #pragma unroll
    for (int r = 0; r < 8; ++r) {
      const int orow = rowTile * 16 + r + 8 * g;
      h[(long)orow * OUT_CH + ct * 16 + m] = acc[ct][r];
    }
  }
}

// --------------------------------------------------------------------------
// Kernel 2: per-node attention scalars aL = h.att[:64], aR = h.att[64:]
// --------------------------------------------------------------------------
__global__ void gat_node_att(const float* __restrict__ h,
                             const float* __restrict__ att,
                             float* __restrict__ aL,
                             float* __restrict__ aR) {
  const int n = blockIdx.x * blockDim.x + threadIdx.x;
  if (n >= N_NODES) return;
  const float4* hr = (const float4*)(h + (long)n * OUT_CH);
  const float4* tL = (const float4*)att;
  const float4* tR = (const float4*)(att + OUT_CH);
  float sL = 0.f, sR = 0.f;
  #pragma unroll
  for (int c = 0; c < OUT_CH / 4; ++c) {
    const float4 v = hr[c], l = tL[c], r = tR[c];
    sL += v.x * l.x + v.y * l.y + v.z * l.z + v.w * l.w;
    sR += v.x * r.x + v.y * r.y + v.z * r.z + v.w * r.w;
  }
  aL[n] = sL;
  aR[n] = sR;
}

// --------------------------------------------------------------------------
// Kernel 3: re-init accumulators every launch (graph replay safety)
// --------------------------------------------------------------------------
__global__ void gat_init(float* __restrict__ acc,
                         unsigned* __restrict__ amaxk,
                         float* __restrict__ esum) {
  const long i = (long)blockIdx.x * blockDim.x + threadIdx.x;
  if (i < (long)N_NODES * OUT_CH) acc[i] = 0.f;
  if (i < N_NODES) { amaxk[i] = 0u; esum[i] = 0.f; }   // key 0 < any encoded float
}

// --------------------------------------------------------------------------
// Kernel 4: per-edge logit + segment max (edges >= N_EDGES are self loops)
// --------------------------------------------------------------------------
__global__ void gat_edge_alpha(const int* __restrict__ ei,
                               const float* __restrict__ aL,
                               const float* __restrict__ aR,
                               float* __restrict__ alpha,
                               unsigned* __restrict__ amaxk) {
  const int e = blockIdx.x * blockDim.x + threadIdx.x;
  if (e >= E_TOT) return;
  int s, d;
  if (e < N_EDGES) { s = ei[e]; d = ei[N_EDGES + e]; }
  else             { s = e - N_EDGES; d = s; }
  float a = aL[d] + aR[s];
  a = (a > 0.f) ? a : NEG_SLOPE * a;
  alpha[e] = a;
  atomicMax(&amaxk[d], fkey(a));
}

// --------------------------------------------------------------------------
// Kernel 5: e = exp(alpha - amax[dst]), segment sum (alpha overwritten by e)
// --------------------------------------------------------------------------
__global__ void gat_edge_exp(const int* __restrict__ ei,
                             float* __restrict__ alpha,
                             const unsigned* __restrict__ amaxk,
                             float* __restrict__ esum) {
  const int e = blockIdx.x * blockDim.x + threadIdx.x;
  if (e >= E_TOT) return;
  const int d = (e < N_EDGES) ? ei[N_EDGES + e] : (e - N_EDGES);
  const float ev = __expf(alpha[e] - funkey(amaxk[d]));
  alpha[e] = ev;
  atomicAdd(&esum[d], ev);
}

// --------------------------------------------------------------------------
// Kernel 6: message scatter: acc[dst] += h[src] * (e / (esum[dst]+1e-16))
// one thread per (edge, 4 channels): b128 gather + 4 f32 atomics, L2-resident
// --------------------------------------------------------------------------
__global__ void gat_scatter(const int* __restrict__ ei,
                            const float* __restrict__ h,
                            const float* __restrict__ alpha,
                            const float* __restrict__ esum,
                            float* __restrict__ acc) {
  const long t = (long)blockIdx.x * blockDim.x + threadIdx.x;
  if (t >= (long)E_TOT * (OUT_CH / 4)) return;
  const int e  = (int)(t >> 4);
  const int c4 = (int)(t & 15) * 4;
  int s, d;
  if (e < N_EDGES) { s = ei[e]; d = ei[N_EDGES + e]; }
  else             { s = e - N_EDGES; d = s; }
  const float coef = alpha[e] / (esum[d] + 1e-16f);
  const float4 hv = *(const float4*)(h + (long)s * OUT_CH + c4);
  float* ap = acc + (long)d * OUT_CH + c4;
  atomicAdd(ap + 0, hv.x * coef);
  atomicAdd(ap + 1, hv.y * coef);
  atomicAdd(ap + 2, hv.z * coef);
  atomicAdd(ap + 3, hv.w * coef);
}

// --------------------------------------------------------------------------
// Kernel 7: out = L2-normalize(acc + bias)  (pure overwrite of d_out)
// --------------------------------------------------------------------------
__global__ void gat_finalize(const float* __restrict__ acc,
                             const float* __restrict__ bias,
                             float* __restrict__ out) {
  const int n = blockIdx.x * blockDim.x + threadIdx.x;
  if (n >= N_NODES) return;
  const float4* r4 = (const float4*)(acc + (long)n * OUT_CH);
  const float4* b4 = (const float4*)bias;
  float4 v[OUT_CH / 4];
  float ss = 0.f;
  #pragma unroll
  for (int c = 0; c < OUT_CH / 4; ++c) {
    float4 t = r4[c]; const float4 b = b4[c];
    t.x += b.x; t.y += b.y; t.z += b.z; t.w += b.w;
    ss += t.x * t.x + t.y * t.y + t.z * t.z + t.w * t.w;
    v[c] = t;
  }
  const float inv = 1.f / fmaxf(sqrtf(ss), 1e-12f);
  float4* o4 = (float4*)(out + (long)n * OUT_CH);
  #pragma unroll
  for (int c = 0; c < OUT_CH / 4; ++c) {
    float4 t = v[c];
    t.x *= inv; t.y *= inv; t.z *= inv; t.w *= inv;
    o4[c] = t;
  }
}

extern "C" void kernel_launch(void* const* d_in, const int* in_sizes, int n_in,
                              void* d_out, int out_size, void* d_ws, size_t ws_size,
                              hipStream_t stream) {
  const float* x    = (const float*)d_in[0];
  const int*   ei   = (const int*)d_in[1];   // [2, N_EDGES]: src row then dst row
  const float* w    = (const float*)d_in[2]; // [128, 64] row-major
  const float* att  = (const float*)d_in[3]; // [128]
  const float* bias = (const float*)d_in[4]; // [64]
  float* out = (float*)d_out;

  char* ws = (char*)d_ws;
  size_t off = 0;
  auto take = [&](size_t bytes) -> char* {
    char* p = ws + off;
    off += (bytes + 255) & ~(size_t)255;
    return p;
  };
  float*    h     = (float*)take((size_t)N_NODES * OUT_CH * sizeof(float));
  float*    acc   = (float*)take((size_t)N_NODES * OUT_CH * sizeof(float));
  unsigned* apk   = (unsigned*)take((size_t)ROW_TILES * KSTEPS * 32 * 8 * sizeof(unsigned));
  unsigned* bpk   = (unsigned*)take((size_t)KSTEPS * 4 * 32 * 8 * sizeof(unsigned));
  float*    aL    = (float*)take((size_t)N_NODES * sizeof(float));
  float*    aR    = (float*)take((size_t)N_NODES * sizeof(float));
  unsigned* amaxk = (unsigned*)take((size_t)N_NODES * sizeof(unsigned));
  float*    esum  = (float*)take((size_t)N_NODES * sizeof(float));
  float*    alpha = (float*)take((size_t)E_TOT * sizeof(float));

  const long na = (long)ROW_TILES * KSTEPS * 32 * 8;
  gat_pack_a<<<(unsigned)((na + 255) / 256), 256, 0, stream>>>(x, apk);
  gat_pack_b<<<(KSTEPS * 4 * 32 * 8 + 255) / 256, 256, 0, stream>>>(w, bpk);
  gat_gemm_bf16<<<(ROW_TILES + 3) / 4, 128, 0, stream>>>(
      (const v16bf*)apk, (const v16bf*)bpk, h);
  gat_node_att<<<(N_NODES + 255) / 256, 256, 0, stream>>>(h, att, aL, aR);
  gat_init<<<(unsigned)(((long)N_NODES * OUT_CH + 255) / 256), 256, 0, stream>>>(acc, amaxk, esum);
  gat_edge_alpha<<<(E_TOT + 255) / 256, 256, 0, stream>>>(ei, aL, aR, alpha, amaxk);
  gat_edge_exp<<<(E_TOT + 255) / 256, 256, 0, stream>>>(ei, alpha, amaxk, esum);
  const long scat = (long)E_TOT * (OUT_CH / 4);
  gat_scatter<<<(unsigned)((scat + 255) / 256), 256, 0, stream>>>(ei, h, alpha, esum, acc);
  gat_finalize<<<(N_NODES + 255) / 256, 256, 0, stream>>>(acc, bias, out);
}
